// NonLocal2D_35862976922210
// MI455X (gfx1250) — matrix operand order repaired and checked by reading, not compiled, not run on hardware
//
#include <hip/hip_runtime.h>
#include <hip/hip_bf16.h>

// ---------------------------------------------------------------------------
// NonLocal2D (SAGAN attention) for MI455X / gfx1250, wave32 + WMMA f16.
// Flash-attention formulation: never materialize the 8x4096x4096 score matrix.
// All big matmuls use v_wmma_f32_16x16x32_f16. B fragments are single 32B
// vector loads; inner loops are software-pipelined with small circular
// buffers sized to stay under 256 live VGPRs (no scratch spills).
// ---------------------------------------------------------------------------

typedef __attribute__((ext_vector_type(16))) _Float16 v16h;
typedef __attribute__((ext_vector_type(8)))  _Float16 v8h;
typedef __attribute__((ext_vector_type(8)))  float    v8f;

#define WMMA_F16(a, b, c) \
  __builtin_amdgcn_wmma_f32_16x16x32_f16(false, (a), false, (b), (short)0, (c), false, false)

#define NB    8           // batch
#define CCH   256         // channels
#define CQ    32          // query/key channels (C/8)
#define NPIX  4096        // H*W
#define NROWS (NB * NPIX) // 32768 total tokens
#define PPITCH 40         // LDS prob-tile pitch (halves); 80B rows keep 16B align

// A-fragment (16x32 f16): per lane 16 contiguous halves at hh*16
// (k = hh*16 + s, matching the contiguous B fragments).
__device__ __forceinline__ v16h load_afrag(const _Float16* rowPtr, int hh) {
  v8h lo = *(const v8h*)(rowPtr + hh * 16);
  v8h hi = *(const v8h*)(rowPtr + hh * 16 + 8);
  return __builtin_shufflevector(lo, hi, 0, 1, 2, 3, 4, 5, 6, 7,
                                 8, 9, 10, 11, 12, 13, 14, 15);
}

// ---------------------------------------------------------------------------
// Spectral normalization: one block per weight. One power iteration exactly
// as the reference (eps=1e-12); writes W/sigma TRANSPOSED as f16
// (WT[j*Cin+i]) so every WMMA B-fragment is one contiguous 32B lane load.
// ---------------------------------------------------------------------------
__global__ void sn_normalize_kernel(const float* __restrict__ W,
                                    const float* __restrict__ u,
                                    int Cout,
                                    _Float16* __restrict__ WT16) {
  const int Cin = CCH;
  __shared__ float sU[CCH], sV[CCH], sRed[CCH];
  const int tid = threadIdx.x;
  sU[tid] = u[tid];
  __syncthreads();

  // v = l2norm(W^T u)
  float vj = 0.f;
  if (tid < Cout) {
    for (int i = 0; i < Cin; ++i) vj += W[i * Cout + tid] * sU[i];
  }
  sRed[tid] = (tid < Cout) ? vj * vj : 0.f;
  __syncthreads();
  for (int s = 128; s > 0; s >>= 1) {
    if (tid < s) sRed[tid] += sRed[tid + s];
    __syncthreads();
  }
  const float nv = sqrtf(sRed[0]);
  sV[tid] = (tid < Cout) ? vj / (nv + 1e-12f) : 0.f;
  __syncthreads();

  // wv = W v ; sigma = ||wv||^2 / (||wv|| + eps)  (== u_new . (W v))
  float wv = 0.f;
  for (int j = 0; j < Cout; ++j) wv += W[tid * Cout + j] * sV[j];
  sRed[tid] = wv * wv;
  __syncthreads();
  for (int s = 128; s > 0; s >>= 1) {
    if (tid < s) sRed[tid] += sRed[tid + s];
    __syncthreads();
  }
  const float nu    = sqrtf(sRed[0]);
  const float sigma = (nu * nu) / (nu + 1e-12f);
  const float inv   = 1.f / sigma;

  for (int idx = tid; idx < Cin * Cout; idx += 256) {
    const int i = idx / Cout, j = idx % Cout;
    WT16[j * Cin + i] = (_Float16)(W[idx] * inv);
  }
}

// ---------------------------------------------------------------------------
// Projections: f = x Wf + bf, g = x Wg + bg, h = x Wh + bh.
// 8 waves/block; each wave owns one 16-row tile, loops 20 column-tile jobs.
// B fragments use a 4-deep circular buffer (load 4 K-steps ahead) so WMMAs
// overlap VMEM without blowing the register budget.
// h is written TRANSPOSED per batch: hT[b][c][n].
// ---------------------------------------------------------------------------
__global__ void proj_kernel(const float* __restrict__ x,
                            const _Float16* __restrict__ WfT, const float* __restrict__ bf,
                            const _Float16* __restrict__ WgT, const float* __restrict__ bg,
                            const _Float16* __restrict__ WhT, const float* __restrict__ bh,
                            _Float16* __restrict__ f16o,   // [NROWS, CQ]
                            _Float16* __restrict__ g16o,   // [NROWS, CQ]
                            _Float16* __restrict__ hT16o)  // [NB, CCH, NPIX]
{
  const int lane    = threadIdx.x & 31;
  const int wid     = blockIdx.x * 8 + (threadIdx.x >> 5);
  const int rowBase = wid * 16;
  const int m  = lane & 15;
  const int hh = lane >> 4;
  const int n  = m;

  // A-fragments for this row tile: two 32B f32 vector loads + cvt per kk.
  v16h aF[8];
#pragma unroll
  for (int kk = 0; kk < 8; ++kk) {
    const float* xr = x + (size_t)(rowBase + m) * CCH + kk * 32;
    v8f lo = *(const v8f*)(xr + hh * 16);
    v8f hi = *(const v8f*)(xr + hh * 16 + 8);
#pragma unroll
    for (int i = 0; i < 8; ++i) {
      aF[kk][i]     = (_Float16)lo[i];
      aF[kk][8 + i] = (_Float16)hi[i];
    }
  }

#pragma unroll 1
  for (int job = 0; job < 20; ++job) {
    const _Float16* WT; const float* bias; int colBase, kind;
    if (job < 2)      { WT = WfT; bias = bf; colBase = job * 16;       kind = 0; }
    else if (job < 4) { WT = WgT; bias = bg; colBase = (job - 2) * 16; kind = 1; }
    else              { WT = WhT; bias = bh; colBase = (job - 4) * 16; kind = 2; }
    const int c = colBase + n;
    const _Float16* wBase = WT + (size_t)c * CCH + hh * 16;

    // 4-deep circular B buffer: 4 loads in flight, 32 live VGPRs.
    v16h bF[4];
#pragma unroll
    for (int p = 0; p < 4; ++p)
      bF[p] = *(const v16h*)(wBase + p * 32);

    v8f acc = {};
#pragma unroll
    for (int kk = 0; kk < 8; ++kk) {
      acc = WMMA_F16(aF[kk], bF[kk & 3], acc);
      if (kk + 4 < 8)
        bF[kk & 3] = *(const v16h*)(wBase + (kk + 4) * 32);
    }

    const float bb = bias[c];
#pragma unroll
    for (int r = 0; r < 8; ++r) {
      const int   q   = rowBase + r + 8 * hh;
      const float val = acc[r] + bb;
      if (kind == 0)      f16o[(size_t)q * CQ + c] = (_Float16)val;
      else if (kind == 1) g16o[(size_t)q * CQ + c] = (_Float16)val;
      else {
        const int bq = q >> 12, nn = q & (NPIX - 1);
        hT16o[(size_t)bq * CCH * NPIX + (size_t)c * NPIX + nn] = (_Float16)val;
      }
    }
  }
}

// ---------------------------------------------------------------------------
// Flash attention: one wave per 16-query tile, online softmax over 4096 keys
// in 32-key blocks. Per block: 2 score WMMAs + 16 o-accum WMMAs.
// Pipelining: bk0/bk1 are reloaded for the NEXT block right after the score
// WMMAs consume them (zero extra registers; loads drain through softmax and
// the 16-WMMA h chain), the h fragment is double-buffered, and the next key
// rows are prefetched (global_prefetch_b8) during the softmax trans ops.
// ---------------------------------------------------------------------------
__global__ void attn_kernel(const _Float16* __restrict__ g16,   // queries [NROWS, CQ]
                            const _Float16* __restrict__ f16k,  // keys    [NROWS, CQ]
                            const _Float16* __restrict__ hT16,  // values  [NB, CCH, NPIX]
                            _Float16* __restrict__ o16)         // [NROWS, CCH]
{
  __shared__ __align__(32) _Float16 pLds[8][16 * PPITCH];
  const int lane  = threadIdx.x & 31;
  const int wloc  = threadIdx.x >> 5;
  const int wid   = blockIdx.x * 8 + wloc;  // 0..2047 query tiles
  const int batch = wid >> 8;               // 256 tiles per batch
  const int qBase = wid * 16;
  const int m  = lane & 15;
  const int hh = lane >> 4;
  const int n  = m;
  _Float16* myP = &pLds[wloc][0];

  // Q fragment (Cq = 32 -> exactly one 16x32 A-fragment).
  const v16h qF = load_afrag(g16 + (size_t)(qBase + m) * CQ, hh);

  v8f acc[16] = {};                 // 16 queries x 256 channels, f32
  float runm[8], runl[8];
#pragma unroll
  for (int r = 0; r < 8; ++r) { runm[r] = -1e30f; runl[r] = 0.f; }

  const _Float16* fB = f16k + (size_t)batch * NPIX * CQ;
  const _Float16* hB = hT16 + (size_t)batch * CCH * NPIX;

  // Preload first key-block score B fragments (B[d][key] = f[key][d];
  // per lane 16 contiguous halves at d = hh*16).
  v16h bk0 = *(const v16h*)(fB + (size_t)(n) * CQ + hh * 16);
  v16h bk1 = *(const v16h*)(fB + (size_t)(16 + n) * CQ + hh * 16);
  const v8f zero = {};

#pragma unroll 1
  for (int kb = 0; kb < NPIX; kb += 32) {
    v8f s0 = WMMA_F16(qF, bk0, zero);
    v8f s1 = WMMA_F16(qF, bk1, zero);

    // bk0/bk1 are dead once the score WMMAs issue: reload them for the next
    // block in place (no extra registers; drains during softmax + h chain).
    const int kbn = (kb + 32) & (NPIX - 1);   // wraps on last iter (harmless)
    bk0 = *(const v16h*)(fB + (size_t)(kbn + n) * CQ + hh * 16);
    bk1 = *(const v16h*)(fB + (size_t)(kbn + 16 + n) * CQ + hh * 16);
    __builtin_prefetch(fB + (size_t)(((kb + 64) & (NPIX - 1)) + n) * CQ, 0, 1);

    // Online softmax over this 32-key block. D layout: row m = r + 8*hh,
    // col = key n; row reductions are shfl_xor across the 16-lane half.
    float scaleA[8];
#pragma unroll
    for (int r = 0; r < 8; ++r) {
      const float x0 = s0[r], x1 = s1[r];
      float rm = fmaxf(x0, x1);
      rm = fmaxf(rm, __shfl_xor(rm, 1, 32));
      rm = fmaxf(rm, __shfl_xor(rm, 2, 32));
      rm = fmaxf(rm, __shfl_xor(rm, 4, 32));
      rm = fmaxf(rm, __shfl_xor(rm, 8, 32));
      const float nm = fmaxf(runm[r], rm);
      const float p0 = __expf(x0 - nm);
      const float p1 = __expf(x1 - nm);
      float rs = p0 + p1;
      rs += __shfl_xor(rs, 1, 32);
      rs += __shfl_xor(rs, 2, 32);
      rs += __shfl_xor(rs, 4, 32);
      rs += __shfl_xor(rs, 8, 32);
      const float sc = __expf(runm[r] - nm);
      runl[r]   = runl[r] * sc + rs;
      runm[r]   = nm;
      scaleA[r] = sc;
      const int mm = r + 8 * hh;
      myP[mm * PPITCH + n]      = (_Float16)p0;  // keys kb..kb+15
      myP[mm * PPITCH + 16 + n] = (_Float16)p1;  // keys kb+16..kb+31
    }

    // Rescale running accumulators (rows match D layout).
#pragma unroll
    for (int t = 0; t < 16; ++t)
#pragma unroll
      for (int r = 0; r < 8; ++r) acc[t][r] *= scaleA[r];

    // Probabilities back as an A-fragment (per-wave LDS is in-order: RAW ok).
    const v16h pA = load_afrag(myP + m * PPITCH, hh);

    // o[16q][256c] += pA(16x32) * h(32x16), 16 channel tiles, double-buffered.
    const _Float16* hRow = hB + kb + hh * 16;
    v16h bcur = *(const v16h*)(hRow + (size_t)n * NPIX);   // t = 0, c = n
#pragma unroll
    for (int t = 0; t < 16; ++t) {
      v16h bnxt;
      if (t < 15)
        bnxt = *(const v16h*)(hRow + (size_t)((t + 1) * 16 + n) * NPIX);
      acc[t] = WMMA_F16(pA, bcur, acc[t]);
      if (t < 15) bcur = bnxt;
    }
  }

  // Normalize and emit o in f16 row-major for the output projection.
#pragma unroll
  for (int t = 0; t < 16; ++t)
#pragma unroll
    for (int r = 0; r < 8; ++r) {
      const int q = qBase + r + 8 * hh;
      const int c = t * 16 + n;
      o16[(size_t)q * CCH + c] = (_Float16)(acc[t][r] / runl[r]);
    }
}

// ---------------------------------------------------------------------------
// Output projection + residual: out = gamma * (o Wc + bc) + x.
// Same 4-deep circular B buffer as proj_kernel.
// ---------------------------------------------------------------------------
__global__ void outproj_kernel(const _Float16* __restrict__ o16,
                               const _Float16* __restrict__ WcT,
                               const float* __restrict__ bc,
                               const float* __restrict__ x,
                               const float* __restrict__ gammaP,
                               float* __restrict__ out) {
  const int lane    = threadIdx.x & 31;
  const int wid     = blockIdx.x * 8 + (threadIdx.x >> 5);
  const int rowBase = wid * 16;
  const int m  = lane & 15;
  const int hh = lane >> 4;
  const int n  = m;
  const float gamma = gammaP[0];

  v16h aF[8];
#pragma unroll
  for (int kk = 0; kk < 8; ++kk)
    aF[kk] = load_afrag(o16 + (size_t)(rowBase + m) * CCH + kk * 32, hh);

#pragma unroll 1
  for (int t = 0; t < 16; ++t) {
    const int c = t * 16 + n;
    const _Float16* wBase = WcT + (size_t)c * CCH + hh * 16;

    v16h bF[4];
#pragma unroll
    for (int p = 0; p < 4; ++p)
      bF[p] = *(const v16h*)(wBase + p * 32);

    v8f acc = {};
#pragma unroll
    for (int kk = 0; kk < 8; ++kk) {
      acc = WMMA_F16(aF[kk], bF[kk & 3], acc);
      if (kk + 4 < 8)
        bF[kk & 3] = *(const v16h*)(wBase + (kk + 4) * 32);
    }

    const float bb = bc[c];
#pragma unroll
    for (int r = 0; r < 8; ++r) {
      const int q = rowBase + r + 8 * hh;
      out[(size_t)q * CCH + c] = gamma * (acc[r] + bb) + x[(size_t)q * CCH + c];
    }
  }
}

// ---------------------------------------------------------------------------
// Launch: 4x spectral norm -> projections -> flash attention -> out proj.
// Workspace (f16): WfT 16K | WgT 16K | WhT 128K | WcT 128K |
//   g 2M | f 2M | hT 16M | o 16M  (~36.3 MB; whole pipeline is L2-resident)
// ---------------------------------------------------------------------------
extern "C" void kernel_launch(void* const* d_in, const int* in_sizes, int n_in,
                              void* d_out, int out_size, void* d_ws, size_t ws_size,
                              hipStream_t stream) {
  const float* x     = (const float*)d_in[0];
  const float* Wf    = (const float*)d_in[1];
  const float* bf    = (const float*)d_in[2];
  const float* uf    = (const float*)d_in[3];
  const float* Wg    = (const float*)d_in[4];
  const float* bg    = (const float*)d_in[5];
  const float* ug    = (const float*)d_in[6];
  const float* Wh    = (const float*)d_in[7];
  const float* bh    = (const float*)d_in[8];
  const float* uh    = (const float*)d_in[9];
  const float* Wc    = (const float*)d_in[10];
  const float* bc    = (const float*)d_in[11];
  const float* uc    = (const float*)d_in[12];
  const float* gamma = (const float*)d_in[13];
  float* out = (float*)d_out;

  char* ws = (char*)d_ws;
  size_t off = 0;
  _Float16* WfT = (_Float16*)(ws + off); off += (size_t)CCH * CQ  * 2;   // 16K
  _Float16* WgT = (_Float16*)(ws + off); off += (size_t)CCH * CQ  * 2;   // 16K
  _Float16* WhT = (_Float16*)(ws + off); off += (size_t)CCH * CCH * 2;   // 128K
  _Float16* WcT = (_Float16*)(ws + off); off += (size_t)CCH * CCH * 2;   // 128K
  _Float16* g16 = (_Float16*)(ws + off); off += (size_t)NROWS * CQ  * 2; // 2M
  _Float16* f16 = (_Float16*)(ws + off); off += (size_t)NROWS * CQ  * 2; // 2M
  _Float16* hT  = (_Float16*)(ws + off); off += (size_t)NROWS * CCH * 2; // 16M
  _Float16* o16 = (_Float16*)(ws + off); off += (size_t)NROWS * CCH * 2; // 16M
  if (ws_size < off) return; // workspace too small; nothing sane to do

  sn_normalize_kernel<<<1, 256, 0, stream>>>(Wf, uf, CQ,  WfT);
  sn_normalize_kernel<<<1, 256, 0, stream>>>(Wg, ug, CQ,  WgT);
  sn_normalize_kernel<<<1, 256, 0, stream>>>(Wh, uh, CCH, WhT);
  sn_normalize_kernel<<<1, 256, 0, stream>>>(Wc, uc, CCH, WcT);

  proj_kernel<<<dim3(NROWS / 16 / 8), dim3(256), 0, stream>>>(
      x, WfT, bf, WgT, bg, WhT, bh, f16, g16, hT);

  attn_kernel<<<dim3(NROWS / 16 / 8), dim3(256), 0, stream>>>(g16, f16, hT, o16);

  outproj_kernel<<<dim3(NROWS / 16 / 8), dim3(256), 0, stream>>>(
      o16, WcT, bc, x, gamma, out);
}
